// CornerPooling_61452392071369
// MI455X (gfx1250) — matrix operand order repaired and checked by reading, not compiled, not run on hardware
//
#include <hip/hip_runtime.h>
#include <stdint.h>

// ---------------------------------------------------------------------------
// CDNA5 / gfx1250 implicit-GEMM corner-pooling block, v2.
// - NHWC bf16 activations with a physical zero halo (no divergent pad guards,
//   tap offsets fold into global_load immediate offsets)
// - v_wmma_f32_16x16x32_bf16, fp32 accumulate, BN folded into weights
// - each wave owns a 16x32 output tile (2 accumulators share one A fragment)
// ---------------------------------------------------------------------------

typedef __attribute__((ext_vector_type(16))) __bf16 v16bf;
typedef __attribute__((ext_vector_type(8)))  __bf16 v8bf;
typedef __attribute__((ext_vector_type(8)))  float  v8f;
typedef __attribute__((ext_vector_type(4)))  int    v4i;

constexpr int NB = 8, H = 128, W = 128, C0 = 256, MIDC = 128;
constexpr int HP = H + 2, WP = W + 2;   // zero-halo padded spatial dims

#define DEVINL static __device__ __forceinline__

DEVINL unsigned short f2bf(float f) {
  unsigned u = __builtin_bit_cast(unsigned, f);
  u += 0x7fffu + ((u >> 16) & 1u);          // round-to-nearest-even
  return (unsigned short)(u >> 16);
}
DEVINL float bf2f(unsigned short h) {
  return __builtin_bit_cast(float, ((unsigned)h) << 16);
}
DEVINL v8f zero8f() {
  v8f a = {0.f, 0.f, 0.f, 0.f, 0.f, 0.f, 0.f, 0.f};
  return a;
}
DEVINL v16bf cat16(v8bf lo, v8bf hi) {
  return __builtin_shufflevector(lo, hi, 0, 1, 2, 3, 4, 5, 6, 7,
                                 8, 9, 10, 11, 12, 13, 14, 15);
}
DEVINL v8f wmma_bf16(v16bf a, v16bf b, v8f c) {
  return __builtin_amdgcn_wmma_f32_16x16x32_bf16(false, a, false, b,
                                                 (short)0, c, false, false);
}
// A fragment: 16(pixels) x 32(channels) from NHWC-bf16; per-lane two 16B loads
// at K-offsets {0..7} and {16..23} (lanes 0-15) / {8..15} and {24..31}.
DEVINL v16bf load_a(const uint16_t* __restrict__ p) {
  return cat16(*(const v8bf*)p, *(const v8bf*)(p + 16));
}

// ---------------------------------------------------------------------------
__global__ void fill0(v4i* __restrict__ p, int n16) {
  int i = blockIdx.x * 256 + threadIdx.x;
  v4i z = {0, 0, 0, 0};
  if (i < n16) p[i] = z;
}

// ---------------------------------------------------------------------------
// NCHW fp32 -> padded NHWC bf16 (interior) via LDS tile transpose
// ---------------------------------------------------------------------------
__global__ __launch_bounds__(256) void to_nhwc(const float* __restrict__ x,
                                               uint16_t* __restrict__ xh) {
  __shared__ float tile[32][33];
  const int tx = threadIdx.x, ty = threadIdx.y;  // (32, 8)
  const int w0 = (blockIdx.x % (W / 32)) * 32;
  const int c0 = (blockIdx.x / (W / 32)) * 32;
  const int h = blockIdx.y, n = blockIdx.z;
  const float* src = x + ((size_t)(n * C0 + c0) * H + h) * W + w0;
#pragma unroll
  for (int k = 0; k < 4; ++k)
    tile[ty + k * 8][tx] = src[(size_t)(ty + k * 8) * H * W + tx];
  __syncthreads();
  uint16_t* dst = xh + ((size_t)(n * HP + h + 1) * WP + (w0 + 1)) * C0 + c0;
#pragma unroll
  for (int k = 0; k < 4; ++k)
    dst[(size_t)(ty + k * 8) * C0 + tx] = f2bf(tile[tx][ty + k * 8]);
}

// ---------------------------------------------------------------------------
// Weight prepack, BN scale folded; layout pw[(tap*KB+kb)][COUT][32] bf16.
// ---------------------------------------------------------------------------
__global__ void prepack(const float* __restrict__ w, const float* __restrict__ g,
                        const float* __restrict__ bta, const float* __restrict__ m,
                        const float* __restrict__ vv, uint16_t* __restrict__ pw,
                        float* __restrict__ bias, int CIN, int COUT, int TAPS) {
  const int idx = blockIdx.x * 256 + threadIdx.x;
  const int KB = CIN / 32;
  if (idx >= TAPS * KB * COUT * 32) return;
  const int kc = idx & 31;
  int t = idx >> 5;
  const int co = t % COUT;
  t /= COUT;
  const int kb = t % KB;
  const int tap = t / KB;
  const int ci = kb * 32 + ((kc & 7) | (((kc >> 4) & 1) << 3) | (((kc >> 3) & 1) << 4));
  const float inv = g[co] * rsqrtf(vv[co] + 1e-5f);
  const size_t widx = (TAPS == 9) ? ((size_t)co * CIN + ci) * 9 + tap
                                  : ((size_t)co * CIN + ci);
  pw[idx] = f2bf(w[widx] * inv);
  if (tap == 0 && kb == 0 && kc == 0) bias[co] = bta[co] - m[co] * inv;
}

// ---------------------------------------------------------------------------
// Fused conv3x3 (C0 -> MIDC) for p1 & p2, bias + ReLU.
// Wave tile: 16 pixels x 32 outch x 2 weight sets = 4 accumulators / 1 A frag.
// Block: 8 waves = 64 pixels x 64 outch.  grid.x = (W/64) * (MIDC/64) = 4.
// ---------------------------------------------------------------------------
__global__ __launch_bounds__(256) void conv_p1p2(
    const uint16_t* __restrict__ xh, const uint16_t* __restrict__ pw1,
    const float* __restrict__ bs1, const uint16_t* __restrict__ pw2,
    const float* __restrict__ bs2, uint16_t* __restrict__ q1,
    uint16_t* __restrict__ q2) {
  constexpr int KB = C0 / 32;  // 8
  const int lane = threadIdx.x & 31, wave = threadIdx.x >> 5;
  const int m = lane & 15;
  const int pix0 = (blockIdx.x & 1) * 64 + (wave & 3) * 16;
  const int ocb = (blockIdx.x >> 1) * 64 + (wave >> 2) * 32;
  const int h = blockIdx.y, n = blockIdx.z;
  v8f a10 = zero8f(), a11 = zero8f(), a20 = zero8f(), a21 = zero8f();
  // per-lane bases; all tap/K offsets become immediates
  const uint16_t* pa =
      xh + ((size_t)(n * HP + h) * WP + pix0 + m) * C0 + ((lane >> 4) << 3);
  const size_t bsel = (size_t)(ocb + m) * 32 + ((lane >> 4) << 4);
#pragma unroll
  for (int dy = 0; dy < 3; ++dy) {
#pragma unroll
    for (int dx = 0; dx < 3; ++dx) {
      const uint16_t* pax = pa + ((size_t)dy * WP + dx) * C0;
      const int tb = (dy * 3 + dx) * KB;
#pragma unroll
      for (int kb = 0; kb < KB; ++kb) {
        const v16bf a = load_a(pax + kb * 32);
        const size_t wo = (size_t)(tb + kb) * MIDC * 32 + bsel;
        a10 = wmma_bf16(a, *(const v16bf*)(pw1 + wo), a10);
        a11 = wmma_bf16(a, *(const v16bf*)(pw1 + wo + 16 * 32), a11);
        a20 = wmma_bf16(a, *(const v16bf*)(pw2 + wo), a20);
        a21 = wmma_bf16(a, *(const v16bf*)(pw2 + wo + 16 * 32), a21);
      }
    }
  }
  const int co0 = ocb + m, co1 = co0 + 16;
  const float b10 = bs1[co0], b11 = bs1[co1], b20 = bs2[co0], b21 = bs2[co1];
  const int mhi = (lane >> 4) << 3;  // C/D rows: M = v + (lane<16 ? 0 : 8)
  uint16_t* o1 =
      q1 + ((size_t)(n * HP + h + 1) * WP + pix0 + mhi + 1) * MIDC;
  uint16_t* o2 = q2 + ((size_t)(n * H + h) * W + pix0 + mhi) * MIDC;
#pragma unroll
  for (int v = 0; v < 8; ++v) {
    float f;
    f = a10[v] + b10; o1[(size_t)v * MIDC + co0] = f2bf(f > 0.f ? f : 0.f);
    f = a11[v] + b11; o1[(size_t)v * MIDC + co1] = f2bf(f > 0.f ? f : 0.f);
    f = a20[v] + b20; o2[(size_t)v * MIDC + co0] = f2bf(f > 0.f ? f : 0.f);
    f = a21[v] + b21; o2[(size_t)v * MIDC + co1] = f2bf(f > 0.f ? f : 0.f);
  }
}

// ---------------------------------------------------------------------------
// Reverse cummax along H (top pool), in place on padded q1 interior.
// ---------------------------------------------------------------------------
__global__ void top_pool_k(uint16_t* __restrict__ q) {
  const int idx = blockIdx.x * 256 + threadIdx.x;
  if (idx >= NB * W * MIDC) return;
  const int c = idx % MIDC;
  const int w = (idx / MIDC) % W;
  const int n = idx / (MIDC * W);
  uint16_t* base = q + ((size_t)(n * HP + 1) * WP + w + 1) * MIDC + c;
  const size_t hs = (size_t)WP * MIDC;
  float run = bf2f(base[(size_t)(H - 1) * hs]);
  for (int hh = H - 2; hh >= 0; --hh) {
    const float vv = bf2f(base[(size_t)hh * hs]);
    run = vv > run ? vv : run;
    base[(size_t)hh * hs] = f2bf(run);
  }
}

// ---------------------------------------------------------------------------
// Reverse cummax along W on q2 (unpadded), fused add into padded q1 interior.
// ---------------------------------------------------------------------------
__global__ void left_pool_add_k(const uint16_t* __restrict__ q2,
                                uint16_t* __restrict__ q1) {
  const int idx = blockIdx.x * 256 + threadIdx.x;
  if (idx >= NB * H * MIDC) return;
  const int c = idx % MIDC;
  const int h = (idx / MIDC) % H;
  const int n = idx / (MIDC * H);
  const uint16_t* b2 = q2 + ((size_t)(n * H + h) * W) * MIDC + c;
  uint16_t* b1 = q1 + ((size_t)(n * HP + h + 1) * WP + 1) * MIDC + c;
  float run = bf2f(b2[(size_t)(W - 1) * MIDC]);
  b1[(size_t)(W - 1) * MIDC] = f2bf(run + bf2f(b1[(size_t)(W - 1) * MIDC]));
  for (int w = W - 2; w >= 0; --w) {
    const float vv = bf2f(b2[(size_t)w * MIDC]);
    run = vv > run ? vv : run;
    b1[(size_t)w * MIDC] = f2bf(run + bf2f(b1[(size_t)w * MIDC]));
  }
}

// ---------------------------------------------------------------------------
// Fused conv3x3(MIDC->C0) on pooled sum + conv1x1(C0->C0) shortcut, ReLU.
// Wave tile 16x32 (2 accumulators).  grid.x = (W/64) * (C0/64) = 8.
// ---------------------------------------------------------------------------
__global__ __launch_bounds__(256) void conv_pool_c1(
    const uint16_t* __restrict__ s, const uint16_t* __restrict__ xh,
    const uint16_t* __restrict__ pwp, const float* __restrict__ bsp,
    const uint16_t* __restrict__ pwc, const float* __restrict__ bsc,
    uint16_t* __restrict__ r1) {
  constexpr int KB1 = MIDC / 32;  // 4
  constexpr int KB2 = C0 / 32;    // 8
  const int lane = threadIdx.x & 31, wave = threadIdx.x >> 5;
  const int m = lane & 15;
  const int pix0 = (blockIdx.x & 1) * 64 + (wave & 3) * 16;
  const int ocb = (blockIdx.x >> 1) * 64 + (wave >> 2) * 32;
  const int h = blockIdx.y, n = blockIdx.z;
  v8f acc0 = zero8f(), acc1 = zero8f();
  const size_t bsel = (size_t)(ocb + m) * 32 + ((lane >> 4) << 4);
  {
    const uint16_t* pa =
        s + ((size_t)(n * HP + h) * WP + pix0 + m) * MIDC + ((lane >> 4) << 3);
#pragma unroll
    for (int dy = 0; dy < 3; ++dy) {
#pragma unroll
      for (int dx = 0; dx < 3; ++dx) {
        const uint16_t* pax = pa + ((size_t)dy * WP + dx) * MIDC;
        const int tb = (dy * 3 + dx) * KB1;
#pragma unroll
        for (int kb = 0; kb < KB1; ++kb) {
          const v16bf a = load_a(pax + kb * 32);
          const size_t wo = (size_t)(tb + kb) * C0 * 32 + bsel;
          acc0 = wmma_bf16(a, *(const v16bf*)(pwp + wo), acc0);
          acc1 = wmma_bf16(a, *(const v16bf*)(pwp + wo + 16 * 32), acc1);
        }
      }
    }
  }
  {  // 1x1 shortcut GEMM into the same accumulators (reads padded interior)
    const uint16_t* pa = xh +
        ((size_t)(n * HP + h + 1) * WP + pix0 + 1 + m) * C0 + ((lane >> 4) << 3);
#pragma unroll
    for (int kb = 0; kb < KB2; ++kb) {
      const v16bf a = load_a(pa + kb * 32);
      const size_t wo = (size_t)kb * C0 * 32 + bsel;
      acc0 = wmma_bf16(a, *(const v16bf*)(pwc + wo), acc0);
      acc1 = wmma_bf16(a, *(const v16bf*)(pwc + wo + 16 * 32), acc1);
    }
  }
  const int co0 = ocb + m, co1 = co0 + 16;
  const float bb0 = bsp[co0] + bsc[co0], bb1 = bsp[co1] + bsc[co1];
  const int mhi = (lane >> 4) << 3;
  uint16_t* o = r1 + ((size_t)(n * HP + h + 1) * WP + pix0 + mhi + 1) * C0;
#pragma unroll
  for (int v = 0; v < 8; ++v) {
    float f;
    f = acc0[v] + bb0; o[(size_t)v * C0 + co0] = f2bf(f > 0.f ? f : 0.f);
    f = acc1[v] + bb1; o[(size_t)v * C0 + co1] = f2bf(f > 0.f ? f : 0.f);
  }
}

// ---------------------------------------------------------------------------
// Final conv3x3 (C0->C0) + bias + ReLU -> fp32 NCHW d_out.
// ---------------------------------------------------------------------------
__global__ __launch_bounds__(256) void conv_c2(const uint16_t* __restrict__ r1,
                                               const uint16_t* __restrict__ pw,
                                               const float* __restrict__ bs,
                                               float* __restrict__ out) {
  constexpr int KB = C0 / 32;  // 8
  const int lane = threadIdx.x & 31, wave = threadIdx.x >> 5;
  const int m = lane & 15;
  const int pix0 = (blockIdx.x & 1) * 64 + (wave & 3) * 16;
  const int ocb = (blockIdx.x >> 1) * 64 + (wave >> 2) * 32;
  const int h = blockIdx.y, n = blockIdx.z;
  v8f acc0 = zero8f(), acc1 = zero8f();
  const uint16_t* pa =
      r1 + ((size_t)(n * HP + h) * WP + pix0 + m) * C0 + ((lane >> 4) << 3);
  const size_t bsel = (size_t)(ocb + m) * 32 + ((lane >> 4) << 4);
#pragma unroll
  for (int dy = 0; dy < 3; ++dy) {
#pragma unroll
    for (int dx = 0; dx < 3; ++dx) {
      const uint16_t* pax = pa + ((size_t)dy * WP + dx) * C0;
      const int tb = (dy * 3 + dx) * KB;
#pragma unroll
      for (int kb = 0; kb < KB; ++kb) {
        const v16bf a = load_a(pax + kb * 32);
        const size_t wo = (size_t)(tb + kb) * C0 * 32 + bsel;
        acc0 = wmma_bf16(a, *(const v16bf*)(pw + wo), acc0);
        acc1 = wmma_bf16(a, *(const v16bf*)(pw + wo + 16 * 32), acc1);
      }
    }
  }
  const int co0 = ocb + m, co1 = co0 + 16;
  const float bb0 = bs[co0], bb1 = bs[co1];
  const int mhi = (lane >> 4) << 3;
  float* o0 = out + ((size_t)(n * C0 + co0) * H + h) * W + pix0 + mhi;
  float* o1 = out + ((size_t)(n * C0 + co1) * H + h) * W + pix0 + mhi;
#pragma unroll
  for (int v = 0; v < 8; ++v) {
    float f;
    f = acc0[v] + bb0; o0[v] = f > 0.f ? f : 0.f;
    f = acc1[v] + bb1; o1[v] = f > 0.f ? f : 0.f;
  }
}

// ---------------------------------------------------------------------------
extern "C" void kernel_launch(void* const* d_in, const int* in_sizes, int n_in,
                              void* d_out, int out_size, void* d_ws,
                              size_t ws_size, hipStream_t stream) {
  (void)in_sizes; (void)n_in; (void)out_size; (void)ws_size;
  const float* x      = (const float*)d_in[0];
  const float* w_p1   = (const float*)d_in[1];
  const float* w_p2   = (const float*)d_in[2];
  const float* w_pool = (const float*)d_in[3];
  const float* w_c1   = (const float*)d_in[4];
  const float* w_c2   = (const float*)d_in[5];
  const float* g_p1 = (const float*)d_in[6],  *b_p1 = (const float*)d_in[7],
             * m_p1 = (const float*)d_in[8],  *v_p1 = (const float*)d_in[9];
  const float* g_p2 = (const float*)d_in[10], *b_p2 = (const float*)d_in[11],
             * m_p2 = (const float*)d_in[12], *v_p2 = (const float*)d_in[13];
  const float* g_pl = (const float*)d_in[14], *b_pl = (const float*)d_in[15],
             * m_pl = (const float*)d_in[16], *v_pl = (const float*)d_in[17];
  const float* g_c1 = (const float*)d_in[18], *b_c1 = (const float*)d_in[19],
             * m_c1 = (const float*)d_in[20], *v_c1 = (const float*)d_in[21];
  const float* g_c2 = (const float*)d_in[22], *b_c2 = (const float*)d_in[23],
             * m_c2 = (const float*)d_in[24], *v_c2 = (const float*)d_in[25];

  char* ws = (char*)d_ws;
  size_t off = 0;
  auto take = [&](size_t bytes) -> char* {
    char* p = ws + off;
    off += (bytes + 255) & ~(size_t)255;
    return p;
  };
  const size_t xh_b = (size_t)NB * HP * WP * C0 * 2;
  const size_t q1_b = (size_t)NB * HP * WP * MIDC * 2;
  const size_t r1_b = (size_t)NB * HP * WP * C0 * 2;
  uint16_t* xh  = (uint16_t*)take(xh_b);
  uint16_t* q1  = (uint16_t*)take(q1_b);
  uint16_t* r1  = (uint16_t*)take(r1_b);
  uint16_t* q2  = (uint16_t*)take((size_t)NB * H * W * MIDC * 2);
  uint16_t* pw1 = (uint16_t*)take((size_t)9 * C0 * MIDC * 2);
  uint16_t* pw2 = (uint16_t*)take((size_t)9 * C0 * MIDC * 2);
  uint16_t* pwp = (uint16_t*)take((size_t)9 * MIDC * C0 * 2);
  uint16_t* pwc = (uint16_t*)take((size_t)C0 * C0 * 2);
  uint16_t* pw2c = (uint16_t*)take((size_t)9 * C0 * C0 * 2);
  float* bs1 = (float*)take(MIDC * 4);
  float* bs2 = (float*)take(MIDC * 4);
  float* bsp = (float*)take(C0 * 4);
  float* bsc = (float*)take(C0 * 4);
  float* bs22 = (float*)take(C0 * 4);

  // 0) zero the padded activation buffers (halo must be 0; interior is
  //    overwritten every call -> deterministic)
  {
    int n16;
    n16 = (int)(xh_b / 16);
    fill0<<<(n16 + 255) / 256, 256, 0, stream>>>((v4i*)xh, n16);
    n16 = (int)(q1_b / 16);
    fill0<<<(n16 + 255) / 256, 256, 0, stream>>>((v4i*)q1, n16);
    n16 = (int)(r1_b / 16);
    fill0<<<(n16 + 255) / 256, 256, 0, stream>>>((v4i*)r1, n16);
  }
  // 1) layout conversion into padded interior
  to_nhwc<<<dim3((W / 32) * (C0 / 32), H, NB), dim3(32, 8), 0, stream>>>(x, xh);
  // 2) weight prepack + BN fold
  {
    int t;
    t = 9 * C0 * MIDC;
    prepack<<<(t + 255) / 256, 256, 0, stream>>>(w_p1, g_p1, b_p1, m_p1, v_p1,
                                                 pw1, bs1, C0, MIDC, 9);
    prepack<<<(t + 255) / 256, 256, 0, stream>>>(w_p2, g_p2, b_p2, m_p2, v_p2,
                                                 pw2, bs2, C0, MIDC, 9);
    t = 9 * MIDC * C0;
    prepack<<<(t + 255) / 256, 256, 0, stream>>>(w_pool, g_pl, b_pl, m_pl, v_pl,
                                                 pwp, bsp, MIDC, C0, 9);
    t = C0 * C0;
    prepack<<<(t + 255) / 256, 256, 0, stream>>>(w_c1, g_c1, b_c1, m_c1, v_c1,
                                                 pwc, bsc, C0, C0, 1);
    t = 9 * C0 * C0;
    prepack<<<(t + 255) / 256, 256, 0, stream>>>(w_c2, g_c2, b_c2, m_c2, v_c2,
                                                 pw2c, bs22, C0, C0, 9);
  }
  // 3) fused branch convs (WMMA)
  conv_p1p2<<<dim3(2 * (MIDC / 64), H, NB), 256, 0, stream>>>(xh, pw1, bs1, pw2,
                                                              bs2, q1, q2);
  // 4) corner pools (q1 := toppool(q1); q1 += leftpool(q2))
  top_pool_k<<<(NB * W * MIDC) / 256, 256, 0, stream>>>(q1);
  left_pool_add_k<<<(NB * H * MIDC) / 256, 256, 0, stream>>>(q2, q1);
  // 5) pool conv + 1x1 shortcut, fused (WMMA)
  conv_pool_c1<<<dim3(2 * (C0 / 64), H, NB), 256, 0, stream>>>(q1, xh, pwp, bsp,
                                                               pwc, bsc, r1);
  // 6) final conv -> fp32 NCHW output (WMMA)
  conv_c2<<<dim3(2 * (C0 / 64), H, NB), 256, 0, stream>>>(r1, pw2c, bs22,
                                                          (float*)d_out);
}
